// MinLSTM_80075370267235
// MI455X (gfx1250) — matrix operand order repaired and checked by reading, not compiled, not run on hardware
//
#include <hip/hip_runtime.h>
#include <stdint.h>

typedef __attribute__((ext_vector_type(16))) _Float16 v16h;
typedef __attribute__((ext_vector_type(8)))  _Float16 v8h;
typedef __attribute__((ext_vector_type(8)))  float    v8f;

// Problem sizes (fixed by the reference)
constexpr int Bb   = 4;
constexpr int Ss   = 4096;
constexpr int Dd   = 1024;
constexpr int Kdim = Dd;        // 1024
constexpr int Ndim = 4 * Dd;    // 4096
constexpr int Mdim = Bb * Ss;   // 16384
constexpr int NCHUNK = 64;
constexpr int CHUNK  = Ss / NCHUNK; // 64

// GEMM LDS staging geometry
constexpr int KSTEP   = 32;
constexpr int LDSPAD  = 40;                 // f16 per row: 32 data + 8 pad (80 B stride)
constexpr int A_ROWS  = 128;
constexpr int B_ROWS  = 256;
constexpr int A_F16   = A_ROWS * LDSPAD;    // 5120
constexpr int B_F16   = B_ROWS * LDSPAD;    // 10240
constexpr int STAGE_F16 = A_F16 + B_F16;    // 15360 f16 = 30720 B per stage

__device__ __forceinline__ float clampg(float x) {
  return fminf(fmaxf(x, -10.0f), 10.0f);
}
__device__ __forceinline__ float softplusf(float z) {
  return log1pf(__expf(z));   // args pre-clipped to [-10,10]
}
__device__ __forceinline__ float sigmoidf(float z) {
  return 1.0f / (1.0f + __expf(-z));
}

// CDNA5 async global->LDS copy (ASYNCcnt-tracked DMA into LDS).
// lds_off = wave-relative LDS byte offset (low 32 bits of generic pointer).
__device__ __forceinline__ void async_copy_b128(uint32_t lds_off, const void* gptr) {
  asm volatile("global_load_async_to_lds_b128 %0, %1, off"
               :
               : "v"(lds_off), "v"(gptr)
               : "memory");
}
__device__ __forceinline__ void wait_asynccnt0() {
  asm volatile("s_wait_asynccnt 0x0" ::: "memory");
}

// ---------------------------------------------------------------------------
// fp32 -> f16 conversion
// ---------------------------------------------------------------------------
__global__ void cvt_f32_to_f16(const float* __restrict__ in,
                               _Float16* __restrict__ out, int n) {
  int i = blockIdx.x * blockDim.x + threadIdx.x;
  if (i < n) out[i] = (_Float16)in[i];
}

// ---------------------------------------------------------------------------
// WMMA GEMM:  out[M,N] = A[M,K] * Bw[N,K]^T + bias[N]
// 256 threads = 8 waves in 2(M) x 4(N); block tile 128x256; 64x64 per wave.
// K-loop double-buffers A/B tiles in LDS via async global->LDS b128 copies.
// ---------------------------------------------------------------------------
__global__ void __launch_bounds__(256)
wmma_gemm_bias(const _Float16* __restrict__ A,
               const _Float16* __restrict__ Bw,
               const float* __restrict__ bias,
               float* __restrict__ out) {
  __shared__ _Float16 lds[2 * STAGE_F16];   // 60 KB

  const int tid   = threadIdx.x;
  const int lane  = tid & 31;
  const int wave  = tid >> 5;
  const int waveM = wave & 1;
  const int waveN = wave >> 1;
  const int mBlock = blockIdx.y * 128;
  const int nBlock = blockIdx.x * 256;
  const int m0    = mBlock + waveM * 64;
  const int n0    = nBlock + waveN * 64;
  const int lhalf = lane >> 4;   // 0: lanes 0-15, 1: lanes 16-31
  const int l16   = lane & 15;

  // Issue one stage of async copies into buffer `buf` for K offset `kt`.
  auto stage_copy = [&](int buf, int kt) {
    const uint32_t base = (uint32_t)(size_t)(&lds[buf * STAGE_F16]);
    // A tile: 128 rows x 64 B  -> 512 b128 transfers (2 per thread)
#pragma unroll
    for (int q = 0; q < 2; ++q) {
      int id  = tid + q * 256;
      int row = id >> 2, seg = id & 3;
      const _Float16* g = A + (size_t)(mBlock + row) * Kdim + kt + seg * 8;
      uint32_t l = base + (uint32_t)(row * LDSPAD + seg * 8) * 2;
      async_copy_b128(l, g);
    }
    // B tile: 256 rows x 64 B -> 1024 b128 transfers (4 per thread)
    const uint32_t bbase = base + (uint32_t)A_F16 * 2;
#pragma unroll
    for (int q = 0; q < 4; ++q) {
      int id  = tid + q * 256;
      int row = id >> 2, seg = id & 3;
      const _Float16* g = Bw + (size_t)(nBlock + row) * Kdim + kt + seg * 8;
      uint32_t l = bbase + (uint32_t)(row * LDSPAD + seg * 8) * 2;
      async_copy_b128(l, g);
    }
  };

  v8f acc[4][4];
#pragma unroll
  for (int i = 0; i < 4; ++i)
#pragma unroll
    for (int j = 0; j < 4; ++j)
#pragma unroll
      for (int e = 0; e < 8; ++e) acc[i][j][e] = 0.0f;

  stage_copy(0, 0);
  wait_asynccnt0();
  __syncthreads();

  int buf = 0;
  for (int kt = 0; kt < Kdim; kt += KSTEP) {
    if (kt + KSTEP < Kdim) stage_copy(buf ^ 1, kt + KSTEP);

    const _Float16* as = &lds[buf * STAGE_F16];
    const _Float16* bs = as + A_F16;

    // Fragment loads from LDS in this code's contiguous k order (same for A and B):
    // A 16x32 f16: lane row = l16; K = lhalf*16 + {0..15} contiguous
    // B 32x16 f16: lane col = l16; K = lhalf*16 + {0..15} contiguous
    v16h af[4], bf[4];
#pragma unroll
    for (int i = 0; i < 4; ++i) {
      const _Float16* ap = as + (waveM * 64 + i * 16 + l16) * LDSPAD + lhalf * 16;
      v8h lo = *(const v8h*)ap;          // K = base+0..7
      v8h hi = *(const v8h*)(ap + 8);    // K = base+8..15
      v16h t;
#pragma unroll
      for (int e = 0; e < 8; ++e) { t[e] = lo[e]; t[e + 8] = hi[e]; }
      af[i] = t;
      bf[i] = *(const v16h*)(bs + (waveN * 64 + i * 16 + l16) * LDSPAD + lhalf * 16);
    }
#pragma unroll
    for (int i = 0; i < 4; ++i)
#pragma unroll
      for (int j = 0; j < 4; ++j)
        acc[i][j] = __builtin_amdgcn_wmma_f32_16x16x32_f16(
            false, af[i], false, bf[j], (short)0, acc[i][j], false, false);

    wait_asynccnt0();     // my copies into back buffer are done
    __syncthreads();      // everyone's copies done & front buffer fully read
    buf ^= 1;
  }

  // Epilogue: C/D layout: VGPR v -> row (v + 8*lhalf), col = l16
#pragma unroll
  for (int j = 0; j < 4; ++j) {
    int col = n0 + j * 16 + l16;
    float bv = bias[col];
#pragma unroll
    for (int i = 0; i < 4; ++i) {
      int rowBase = m0 + i * 16 + lhalf * 8;
#pragma unroll
      for (int v = 0; v < 8; ++v)
        out[(size_t)(rowBase + v) * Ndim + col] = acc[i][j][v] + bv;
    }
  }
}

// ---------------------------------------------------------------------------
// Chunked linear scan:  c_t = f'_t * c_{t-1} + i'_t * g(cell_t)
// gates layout (B, S, 4D): [i | f | o | cell] each of width D.
// ---------------------------------------------------------------------------
__global__ void scan_pass1(const float* __restrict__ gates,
                           float* __restrict__ chunkF,
                           float* __restrict__ chunkV) {
  int idx   = blockIdx.x * blockDim.x + threadIdx.x;
  int d     = idx & (Dd - 1);
  int t     = idx >> 10;           // Dd = 1024
  int chunk = t & (NCHUNK - 1);
  int b     = t >> 6;              // NCHUNK = 64
  const float* gb = gates + (size_t)b * Ss * Ndim;
  float F = 1.0f, V = 0.0f;
  int s0 = chunk * CHUNK;
  for (int s = s0; s < s0 + CHUNK; ++s) {
    const float* g = gb + (size_t)s * Ndim;
    float ig = clampg(g[d]);
    float fg = clampg(g[Dd + d]);
    float cg = clampg(g[3 * Dd + d]);
    float diff = softplusf(-fg) - softplusf(-ig);
    float fp = sigmoidf(-diff);          // f' = sigmoid(-diff)
    float ip = 1.0f - fp;                // i' = sigmoid(diff)
    float gv = (cg >= 0.0f) ? (cg + 0.5f) : sigmoidf(cg);
    F = fp * F;
    V = fp * V + ip * gv;
  }
  size_t o = ((size_t)chunk * Bb + b) * Dd + d;
  chunkF[o] = F;
  chunkV[o] = V;
}

__global__ void scan_pass2(const float* __restrict__ chunkF,
                           const float* __restrict__ chunkV,
                           const float* __restrict__ c_prev,  // (B,1,D)
                           float* __restrict__ cEntry) {
  int idx = blockIdx.x * blockDim.x + threadIdx.x;  // 0 .. B*D-1
  int d = idx & (Dd - 1);
  int b = idx >> 10;
  float c = c_prev[(size_t)b * Dd + d];
  for (int ch = 0; ch < NCHUNK; ++ch) {
    size_t o = ((size_t)ch * Bb + b) * Dd + d;
    cEntry[o] = c;
    c = chunkF[o] * c + chunkV[o];
  }
}

template <bool F16OUT>
__global__ void scan_pass3(const float* __restrict__ gates,
                           const float* __restrict__ cEntry,
                           void* __restrict__ hout) {
  int idx   = blockIdx.x * blockDim.x + threadIdx.x;
  int d     = idx & (Dd - 1);
  int t     = idx >> 10;
  int chunk = t & (NCHUNK - 1);
  int b     = t >> 6;
  const float* gb = gates + (size_t)b * Ss * Ndim;
  float c = cEntry[((size_t)chunk * Bb + b) * Dd + d];
  int s0 = chunk * CHUNK;
  for (int s = s0; s < s0 + CHUNK; ++s) {
    const float* g = gb + (size_t)s * Ndim;
    float ig = clampg(g[d]);
    float fg = clampg(g[Dd + d]);
    float og = clampg(g[2 * Dd + d]);
    float cg = clampg(g[3 * Dd + d]);
    float diff = softplusf(-fg) - softplusf(-ig);
    float fp = sigmoidf(-diff);
    float ip = 1.0f - fp;
    float gv = (cg >= 0.0f) ? (cg + 0.5f) : sigmoidf(cg);
    c = fp * c + ip * gv;
    float h = sigmoidf(og) * c;
    size_t ho = ((size_t)b * Ss + s) * Dd + d;
    if (F16OUT) ((_Float16*)hout)[ho] = (_Float16)h;
    else        ((float*)hout)[ho]    = h;
  }
}

// ---------------------------------------------------------------------------
// Launch
// ---------------------------------------------------------------------------
extern "C" void kernel_launch(void* const* d_in, const int* in_sizes, int n_in,
                              void* d_out, int out_size, void* d_ws, size_t ws_size,
                              hipStream_t stream) {
  const float* x   = (const float*)d_in[0];
  const float* W0  = (const float*)d_in[1];
  const float* b0  = (const float*)d_in[2];
  const float* W1  = (const float*)d_in[3];
  const float* b1  = (const float*)d_in[4];
  const float* c0p = (const float*)d_in[5];
  const float* c1p = (const float*)d_in[6];
  float* out = (float*)d_out;

  char* ws = (char*)d_ws;
  size_t off = 0;
  auto walloc = [&](size_t bytes) -> void* {
    void* p = ws + off;
    off += (bytes + 255) & ~(size_t)255;
    return p;
  };
  float*    gates  = (float*)   walloc((size_t)Mdim * Ndim * sizeof(float));     // 256 MB, reused
  _Float16* x16    = (_Float16*)walloc((size_t)Mdim * Kdim * sizeof(_Float16));  // 32 MB
  _Float16* h16    = (_Float16*)walloc((size_t)Mdim * Kdim * sizeof(_Float16));  // 32 MB
  _Float16* w016   = (_Float16*)walloc((size_t)Ndim * Kdim * sizeof(_Float16));  // 8 MB
  _Float16* w116   = (_Float16*)walloc((size_t)Ndim * Kdim * sizeof(_Float16));  // 8 MB
  float*    chunkF = (float*)   walloc((size_t)NCHUNK * Bb * Dd * sizeof(float));
  float*    chunkV = (float*)   walloc((size_t)NCHUNK * Bb * Dd * sizeof(float));
  float*    cEntry = (float*)   walloc((size_t)NCHUNK * Bb * Dd * sizeof(float));

  // --- precision conversion ---
  const int nx = Mdim * Kdim;
  const int nw = Ndim * Kdim;
  cvt_f32_to_f16<<<(nx + 255) / 256, 256, 0, stream>>>(x,  x16,  nx);
  cvt_f32_to_f16<<<(nw + 255) / 256, 256, 0, stream>>>(W0, w016, nw);
  cvt_f32_to_f16<<<(nw + 255) / 256, 256, 0, stream>>>(W1, w116, nw);

  const dim3 gemmGrid(Ndim / 256, Mdim / 128);
  const int scanThreads = Bb * NCHUNK * Dd;         // 262144
  const int scanBlocks  = scanThreads / 256;
  const int chanBlocks  = (Bb * Dd) / 256;

  // --- layer 1 ---
  wmma_gemm_bias<<<gemmGrid, 256, 0, stream>>>(x16, w016, b0, gates);
  scan_pass1<<<scanBlocks, 256, 0, stream>>>(gates, chunkF, chunkV);
  scan_pass2<<<chanBlocks, 256, 0, stream>>>(chunkF, chunkV, c0p, cEntry);
  scan_pass3<true><<<scanBlocks, 256, 0, stream>>>(gates, cEntry, h16);

  // --- layer 2 ---
  wmma_gemm_bias<<<gemmGrid, 256, 0, stream>>>(h16, w116, b1, gates);
  scan_pass1<<<scanBlocks, 256, 0, stream>>>(gates, chunkF, chunkV);
  scan_pass2<<<chanBlocks, 256, 0, stream>>>(chunkF, chunkV, c1p, cEntry);
  scan_pass3<false><<<scanBlocks, 256, 0, stream>>>(gates, cEntry, out);
}
